// DCE_42202348650598
// MI455X (gfx1250) — matrix operand order repaired and checked by reading, not compiled, not run on hardware
//
#include <hip/hip_runtime.h>
#include <stdint.h>

// ---------------------------------------------------------------------------
// CDNA5 / gfx1250 implementation.
// Dominant math done with v_wmma_f32_16x16x32_bf16 (wave32 WMMA).
// B-panel staging uses GLOBAL_LOAD_ASYNC_TO_LDS_B128 (ASYNCcnt) when the
// toolchain declares the builtin; otherwise falls back to load/ds_store.
// ---------------------------------------------------------------------------

typedef __attribute__((ext_vector_type(16))) __bf16 bf16x16;
typedef __attribute__((ext_vector_type(8)))  __bf16 bf16x8;
typedef __attribute__((ext_vector_type(8)))  float  f32x8;
typedef int v4i __attribute__((vector_size(16)));   // int4, matches builtin param

#define REF_EPS 1e-7f
#define NEG_BIG (-3.4e38f)

// Sizes for this problem instance.
#define BB 4
#define NN 5
#define CC 256
#define SS 1024   // 32*32
#define XX (BB*NN)          // 20
#define OC 256              // output channels
#define IC 512              // 2*c

// ---- fragment helpers ------------------------------------------------------
// A-operand (16x32 bf16, M x K): lane m = lane&15; kb8 = (lane>=16)?8:0.
//   VGPR0-3 hold K = kb8..kb8+7 ; VGPR4-7 hold K = 16+kb8 .. 16+kb8+7.
// B-operand (32x16 bf16, K x N): lane n = lane&15; kb8 = (lane>=16)?8:0, as for A.
//   VGPR0-3 hold K = kb8..kb8+7 ; VGPR4-7 hold K = 16+kb8 .. 16+kb8+7.
static __device__ __forceinline__ bf16x16 make_frag(const __bf16* p0, const __bf16* p1) {
    bf16x8 lo = *(const bf16x8*)p0;
    bf16x8 hi = *(const bf16x8*)p1;
    return __builtin_shufflevector(lo, hi, 0,1,2,3,4,5,6,7,8,9,10,11,12,13,14,15);
}

static __device__ __forceinline__ f32x8 wmma_bf16(bf16x16 a, bf16x16 b, f32x8 c) {
    // (neg_a, A, neg_b, B, c_mod, C, reuse_a, reuse_b)
    return __builtin_amdgcn_wmma_f32_16x16x32_bf16(false, a, false, b, (short)0, c, false, false);
}

// ---------------------------------------------------------------------------
// Phase A: per-column L2 norms over channels, and normalized transposed bf16
// copy qhatT[x][s][c] = q[x][c][s] / ||q[x][:,s]||   (c contiguous).
// grid = XX*4 blocks of 256 threads (each thread = one spatial column).
// ---------------------------------------------------------------------------
__global__ __launch_bounds__(256) void normalize_kernel(const float* __restrict__ q,
                                                        float* __restrict__ norms,
                                                        __bf16* __restrict__ qhatT) {
    int x = blockIdx.x >> 2;
    int p = ((blockIdx.x & 3) << 8) + threadIdx.x;
    const float* base = q + (size_t)x * CC * SS + p;
    float s = 0.f;
    for (int c = 0; c < CC; ++c) { float v = base[(size_t)c * SS]; s += v * v; }
    float nrm = sqrtf(s);
    norms[(size_t)x * SS + p] = nrm;
    float inv = 1.0f / fmaxf(nrm, 1e-20f);
    __bf16* dst = qhatT + ((size_t)x * SS + p) * CC;
    for (int c = 0; c < CC; ++c) dst[c] = (__bf16)(base[(size_t)c * SS] * inv);
}

// ---------------------------------------------------------------------------
// W (256 x 512, f32 row-major) -> bf16 row-major (same layout).
// ---------------------------------------------------------------------------
__global__ __launch_bounds__(256) void wcvt_kernel(const float* __restrict__ W,
                                                   __bf16* __restrict__ Wbf) {
    int i = blockIdx.x * 256 + threadIdx.x;
    if (i < OC * IC) Wbf[i] = (__bf16)W[i];
}

// ---------------------------------------------------------------------------
// Phase B: maxsim[b,i,j,p] = max_s  qhat[b,j+1,:,s] . qhat[b,i,:,p]
// block = (bi, j, p-tile of 64).  8 waves; wave w covers s in [128w,128w+128).
// B panel (64 rows x 256 ch bf16) staged in LDS with padded rows (bank spread).
// Each 16-s step: 8x v_wmma_f32_16x16x32_bf16 per 16x16 tile (K=256), 4 tiles.
// ---------------------------------------------------------------------------
#define LDSROW 264   // 256 + 8 bf16 pad -> 528B row stride, 16B aligned
__global__ __launch_bounds__(256) void simmax_kernel(const __bf16* __restrict__ qhatT,
                                                     float* __restrict__ maxsim) {
    __shared__ __bf16 ldsB[64 * LDSROW];
    __shared__ float  maxwave[8][64];

    int bid = blockIdx.x;
    int pt  = bid & 15;          // p-tile (64 wide)
    int j   = (bid >> 4) & 3;    // 0..3   (qs index)
    int bi  = bid >> 6;          // 0..19  (= b*5 + i, qf index)
    int b   = bi / NN;
    int p0  = pt * 64;
    int tid = threadIdx.x;

    // stage B panel: rows p0..p0+63, 256 bf16 each (contiguous in qhatT)
    const __bf16* Bg = qhatT + ((size_t)bi * SS + p0) * CC;
#if __has_builtin(__builtin_amdgcn_global_load_async_to_lds_b128)
    #pragma unroll
    for (int it = 0; it < 8; ++it) {
        int idx = tid + it * 256;       // 64 rows x 32 chunks of 16 bytes
        int row = idx >> 5, col = idx & 31;
        __builtin_amdgcn_global_load_async_to_lds_b128(
            (__attribute__((address_space(1))) v4i*)&Bg[(size_t)row * CC + col * 8],
            (__attribute__((address_space(3))) v4i*)&ldsB[row * LDSROW + col * 8],
            0, 0);
    }
    asm volatile("s_wait_asynccnt 0x0" ::: "memory");
#else
    #pragma unroll
    for (int it = 0; it < 8; ++it) {
        int idx = tid + it * 256;       // 64 rows x 32 chunks of 8 bf16
        int row = idx >> 5, col = idx & 31;
        *(bf16x8*)&ldsB[row * LDSROW + col * 8] =
            *(const bf16x8*)&Bg[(size_t)row * CC + col * 8];
    }
#endif
    __syncthreads();

    int wave = tid >> 5, lane = tid & 31, ln = lane & 15, up = lane >> 4;
    int kb8 = up * 8, kb16 = up * 16;
    const __bf16* Abase = qhatT + (size_t)(b * NN + 1 + j) * SS * CC;

    float rm[4] = {NEG_BIG, NEG_BIG, NEG_BIG, NEG_BIG};

    for (int st = 0; st < 8; ++st) {               // 16-row s-tiles in this wave's chunk
        int s0 = wave * 128 + st * 16;
        const __bf16* Arow = Abase + (size_t)(s0 + ln) * CC;
        f32x8 acc[4] = {};
        #pragma unroll
        for (int kk = 0; kk < 8; ++kk) {           // K = 256 in steps of 32
            bf16x16 a = make_frag(Arow + kk * 32 + kb8, Arow + kk * 32 + 16 + kb8);
            #pragma unroll
            for (int t = 0; t < 4; ++t) {
                const __bf16* Bp = &ldsB[(t * 16 + ln) * LDSROW + kk * 32 + kb8];
                bf16x16 bf = make_frag(Bp, Bp + 16);
                acc[t] = wmma_bf16(a, bf, acc[t]);
            }
        }
        #pragma unroll
        for (int t = 0; t < 4; ++t) {              // fold max over the 16 s-rows
            float m = acc[t][0];
            #pragma unroll
            for (int r = 1; r < 8; ++r) m = fmaxf(m, acc[t][r]);
            rm[t] = fmaxf(rm[t], m);
        }
    }
    #pragma unroll
    for (int t = 0; t < 4; ++t) {                  // combine M-halves (wave32)
        rm[t] = fmaxf(rm[t], __shfl_xor(rm[t], 16, 32));
        if (lane < 16) maxwave[wave][t * 16 + ln] = rm[t];
    }
    __syncthreads();
    if (tid < 64) {                                // combine the 8 s-chunks
        float v = maxwave[0][tid];
        #pragma unroll
        for (int w = 1; w < 8; ++w) v = fmaxf(v, maxwave[w][tid]);
        maxsim[((size_t)bi * 4 + j) * SS + p0 + tid] = v;
    }
}

// ---------------------------------------------------------------------------
// Phase C: per (b,i): per-j min/max over p, then att = mean_j minmax-norm.
// ---------------------------------------------------------------------------
__global__ __launch_bounds__(256) void att_kernel(const float* __restrict__ maxsim,
                                                  float* __restrict__ att) {
    __shared__ float smn[8], smx[8];
    __shared__ float jmn[4], jmx[4];
    int bi = blockIdx.x;
    int tid = threadIdx.x, wave = tid >> 5, lane = tid & 31;
    for (int j = 0; j < 4; ++j) {
        const float* m = maxsim + ((size_t)bi * 4 + j) * SS;
        float mn = 3.4e38f, mx = NEG_BIG;
        #pragma unroll
        for (int k = 0; k < 4; ++k) {
            float v = m[tid + k * 256];
            mn = fminf(mn, v); mx = fmaxf(mx, v);
        }
        #pragma unroll
        for (int o = 16; o >= 1; o >>= 1) {
            mn = fminf(mn, __shfl_xor(mn, o, 32));
            mx = fmaxf(mx, __shfl_xor(mx, o, 32));
        }
        if (lane == 0) { smn[wave] = mn; smx[wave] = mx; }
        __syncthreads();
        if (tid == 0) {
            float a = smn[0], c = smx[0];
            #pragma unroll
            for (int w = 1; w < 8; ++w) { a = fminf(a, smn[w]); c = fmaxf(c, smx[w]); }
            jmn[j] = a; jmx[j] = c;
        }
        __syncthreads();
    }
    #pragma unroll
    for (int k = 0; k < 4; ++k) {
        int p = tid + k * 256;
        float s = 0.f;
        #pragma unroll
        for (int j = 0; j < 4; ++j) {
            float v = maxsim[((size_t)bi * 4 + j) * SS + p];
            s += (v - jmn[j]) / (jmx[j] - jmn[j] + REF_EPS);
        }
        att[(size_t)bi * SS + p] = 0.25f * s;
    }
}

// ---------------------------------------------------------------------------
// Phase D: out[x,o,s] = qn[x,s]*(W1.qhat + att[x,s]*W2.qhat) + bias[o]
// block = (x, s-tile of 128). Wave holds its 16-s B fragments (K=256) in
// registers; streams W fragments; dual accumulators for W1/W2.
// ---------------------------------------------------------------------------
__global__ __launch_bounds__(256) void out_kernel(const __bf16* __restrict__ qhatT,
                                                  const float* __restrict__ norms,
                                                  const float* __restrict__ att,
                                                  const __bf16* __restrict__ Wbf,
                                                  const float* __restrict__ bias,
                                                  float* __restrict__ out) {
    int bid = blockIdx.x;
    int x = bid >> 3, stile = bid & 7;
    int tid = threadIdx.x, wave = tid >> 5, lane = tid & 31, ln = lane & 15, up = lane >> 4;
    int kb8 = up * 8, kb16 = up * 16;
    int s_l = stile * 128 + wave * 16 + ln;

    const __bf16* Brow = qhatT + ((size_t)x * SS + s_l) * CC;
    bf16x16 Bf[8];
    #pragma unroll
    for (int kk = 0; kk < 8; ++kk)
        Bf[kk] = make_frag(Brow + kk * 32 + kb8, Brow + kk * 32 + kb8 + 16);

    float att_s = att[(size_t)x * SS + s_l];
    float qn_s  = norms[(size_t)x * SS + s_l];

    for (int ot = 0; ot < 16; ++ot) {              // 16 o-tiles of 16 rows
        f32x8 acc1 = {}, acc2 = {};
        const __bf16* Wr = Wbf + (size_t)(ot * 16 + ln) * IC;
        #pragma unroll
        for (int kk = 0; kk < 8; ++kk) {
            bf16x16 w1 = make_frag(Wr + kk * 32 + kb8,        Wr + kk * 32 + kb8 + 16);
            bf16x16 w2 = make_frag(Wr + CC + kk * 32 + kb8,   Wr + CC + kk * 32 + kb8 + 16);
            acc1 = wmma_bf16(w1, Bf[kk], acc1);
            acc2 = wmma_bf16(w2, Bf[kk], acc2);
        }
        #pragma unroll
        for (int r = 0; r < 8; ++r) {
            int o = ot * 16 + up * 8 + r;
            out[((size_t)x * OC + o) * SS + s_l] =
                qn_s * (acc1[r] + att_s * acc2[r]) + bias[o];
        }
    }
}

// ---------------------------------------------------------------------------
// Workspace layout (bytes, 256-aligned):
//   qhatT  bf16  : XX*SS*CC*2     = 10,485,760
//   norms  f32   : XX*SS*4        =     81,920
//   Wbf    bf16  : OC*IC*2        =    262,144
//   maxsim f32   : XX*4*SS*4      =    327,680
//   att    f32   : XX*SS*4        =     81,920
// total ~11.24 MB
// ---------------------------------------------------------------------------
#define OFF_QHAT   ((size_t)0)
#define OFF_NORMS  ((size_t)10485760)
#define OFF_WBF    ((size_t)10567680)
#define OFF_MAXSIM ((size_t)10829824)
#define OFF_ATT    ((size_t)11157504)
#define WS_NEEDED  ((size_t)11239424)

extern "C" void kernel_launch(void* const* d_in, const int* in_sizes, int n_in,
                              void* d_out, int out_size, void* d_ws, size_t ws_size,
                              hipStream_t stream) {
    (void)in_sizes; (void)n_in; (void)out_size;
    if (ws_size < WS_NEEDED) return;

    const float* querys = (const float*)d_in[0];
    const float* W      = (const float*)d_in[1];
    const float* bias   = (const float*)d_in[2];
    float* out = (float*)d_out;

    char* ws = (char*)d_ws;
    __bf16* qhatT  = (__bf16*)(ws + OFF_QHAT);
    float*  norms  = (float*) (ws + OFF_NORMS);
    __bf16* Wbf    = (__bf16*)(ws + OFF_WBF);
    float*  maxsim = (float*) (ws + OFF_MAXSIM);
    float*  att    = (float*) (ws + OFF_ATT);

    normalize_kernel<<<XX * 4, 256, 0, stream>>>(querys, norms, qhatT);
    wcvt_kernel<<<(OC * IC + 255) / 256, 256, 0, stream>>>(W, Wbf);
    simmax_kernel<<<BB * NN * 4 * 16, 256, 0, stream>>>(qhatT, maxsim);
    att_kernel<<<XX, 256, 0, stream>>>(maxsim, att);
    out_kernel<<<XX * 8, 256, 0, stream>>>(qhatT, norms, att, Wbf, bias, out);
}
